// BahdanauAttention_28338194219009
// MI455X (gfx1250) — matrix operand order repaired and checked by reading, not compiled, not run on hardware
//
#include <hip/hip_runtime.h>
#include <hip/hip_bf16.h>

// Problem dims (compile-time constants from the reference)
#define Bdim 8
#define Tdim 128
#define Sdim 256
#define Hdim 512

#define KC    64            // K-chunk staged in LDS per block iteration
#define LPITCH 72           // LDS row pitch in bf16 (64 + 8 pad: conflict-free)

typedef __bf16  v16bf __attribute__((ext_vector_type(16)));
typedef __bf16  v8bf  __attribute__((ext_vector_type(8)));
typedef __bf16  v4bf  __attribute__((ext_vector_type(4)));
typedef float   v8f   __attribute__((ext_vector_type(8)));

// float -> bf16 with round-to-nearest-even (bit-exact, no libcall dependence)
static __device__ __forceinline__ __bf16 f2bf(float f) {
    unsigned u = __builtin_bit_cast(unsigned, f);
    u += 0x7FFFu + ((u >> 16) & 1u);
    unsigned short hs = (unsigned short)(u >> 16);
    return __builtin_bit_cast(__bf16, hs);
}

// -----------------------------------------------------------------------------
// Kernel 0: packed fp32 -> bf16 conversion (each element converted exactly once)
// -----------------------------------------------------------------------------
__global__ void cvt_f32_to_bf16(const float* __restrict__ src,
                                __bf16* __restrict__ dst, int n4) {
    const int i = blockIdx.x * blockDim.x + threadIdx.x;
    if (i < n4) {
        const float4 f = ((const float4*)src)[i];
        v4bf o;
        o[0] = f2bf(f.x); o[1] = f2bf(f.y); o[2] = f2bf(f.z); o[3] = f2bf(f.w);
        ((v4bf*)dst)[i] = o;
    }
}

// -----------------------------------------------------------------------------
// Kernel 1: out[m,n] = sum_k X[m,k]*W[n,k] + bias[n], bf16 in, fp32 accumulate.
//   Block = 4 waves = 64x64 output tile. Per K-chunk (KC=64):
//     * 128 threads cooperatively stage X(64xKC) and W(64xKC) into padded LDS
//       with fully coalesced global_load_b128s (each element fetched once per
//       block instead of once per wave -> 4x less B traffic, coalesced A).
//     * prefetch next chunk (global_prefetch_b8) while WMMAs run.
//     * fragments come from LDS (ds_load_b128), per ISA 7.12.2 layouts:
//         A: lane {half,l15}, row M=l15, elems = k2+half*8+0..7 | +16
//         B: lane {half,l15}, col N=l15, elems = k2+half*8+0..7 | +16
//         C/D: VGPR r holds M = half*8 + r, N = l15
//   Wave wid owns M-tile wid and all 4 N-tiles (A fragment reused 4x).
// -----------------------------------------------------------------------------
__global__ __launch_bounds__(128)
void proj_gemm_bf16_wmma(const __bf16* __restrict__ X,   // M x 512 bf16
                         const __bf16* __restrict__ W,   // 512 x 512 bf16
                         const float*  __restrict__ bias,
                         float* __restrict__ out,        // M x 512 fp32
                         int M) {
    __shared__ __bf16 Xs[64][LPITCH];
    __shared__ __bf16 Ws[64][LPITCH];

    const int lane = threadIdx.x;            // 0..31 (wave32)
    const int half = lane >> 4;              // 0 or 1
    const int l15  = lane & 15;
    const int wid  = threadIdx.y;            // wave id 0..3 -> M-tile
    const int li   = wid * 32 + lane;        // 0..127 linear id

    const int mBase = blockIdx.y * 64;
    const int nBase = blockIdx.x * 64;

    // Staging map: group g (0..511) = 8 bf16; row = g>>3, col = (g&7)*8.
    const int srow = li >> 3;                // rows 0..15 per j-step of 16
    const int scol = (li & 7) * 8;

    v8f c[4];
    #pragma unroll
    for (int nt = 0; nt < 4; ++nt)
        #pragma unroll
        for (int r = 0; r < 8; ++r) c[nt][r] = 0.0f;

    for (int kk = 0; kk < Hdim; kk += KC) {
        // ---- cooperative staging: 4 b128 per thread per matrix ------------
        #pragma unroll
        for (int j = 0; j < 4; ++j) {
            const int row = srow + j * 16;
            *(v8bf*)&Xs[row][scol] =
                *(const v8bf*)(X + (size_t)(mBase + row) * Hdim + kk + scol);
            *(v8bf*)&Ws[row][scol] =
                *(const v8bf*)(W + (size_t)(nBase + row) * Hdim + kk + scol);
        }
        // Prefetch the next chunk into cache while this one computes.
        if (kk + KC < Hdim) {
            __builtin_prefetch(X + (size_t)(mBase + srow) * Hdim + kk + KC + scol, 0, 0);
            __builtin_prefetch(W + (size_t)(nBase + srow) * Hdim + kk + KC + scol, 0, 0);
        }
        __syncthreads();

        // ---- WMMA over the staged chunk -----------------------------------
        #pragma unroll
        for (int k2 = 0; k2 < KC; k2 += 32) {
            const int ar = wid * 16 + l15;
            const v8bf a_lo = *(const v8bf*)&Xs[ar][k2 + half * 8];
            const v8bf a_hi = *(const v8bf*)&Xs[ar][k2 + 16 + half * 8];
            const v16bf a = __builtin_shufflevector(a_lo, a_hi,
                    0, 1, 2, 3, 4, 5, 6, 7, 8, 9, 10, 11, 12, 13, 14, 15);
            #pragma unroll
            for (int nt = 0; nt < 4; ++nt) {
                const v8bf b_lo = *(const v8bf*)&Ws[nt * 16 + l15][k2 + half * 8];
                const v8bf b_hi = *(const v8bf*)&Ws[nt * 16 + l15][k2 + 16 + half * 8];
                const v16bf b = __builtin_shufflevector(b_lo, b_hi,
                    0, 1, 2, 3, 4, 5, 6, 7, 8, 9, 10, 11, 12, 13, 14, 15);
                // 8 args: (neg_a, A, neg_b, B, c_mod, C, reuse_a, reuse_b)
                c[nt] = __builtin_amdgcn_wmma_f32_16x16x32_bf16(
                            false, a, false, b, (short)0, c[nt], false, false);
            }
        }
        __syncthreads();
    }

    // ---- epilogue: bias add, fp32 store -----------------------------------
    #pragma unroll
    for (int nt = 0; nt < 4; ++nt) {
        const int ng = nBase + nt * 16 + l15;
        const float bn = bias[ng];
        #pragma unroll
        for (int r = 0; r < 8; ++r) {
            const int mg = mBase + wid * 16 + (half << 3) + r;
            out[(size_t)mg * Hdim + ng] = c[nt][r] + bn;
        }
    }
}

// -----------------------------------------------------------------------------
// Kernel 2: fused score + masked softmax + context, one block per (b,t).
//   score[s] = sum_h v[h]*tanh(qproj[b,t,h] + hproj[b,s,h]) + v_b   (thread s)
//   softmax over s with length mask; context[b,t,h] = sum_s attn[s]*enc[b,s,h]
// The (B,T,S,H) intermediate never touches memory (would be 537 MB).
// -----------------------------------------------------------------------------
__global__ __launch_bounds__(Sdim)
void attn_fused(const float* __restrict__ qproj,   // (B*T, H)
                const float* __restrict__ hproj,   // (B*S, H)
                const float* __restrict__ enc,     // (B, S, H)
                const int*   __restrict__ src_len, // (B,)
                const float* __restrict__ v_w,     // (H,)
                const float* __restrict__ v_b,     // (1,)
                float* __restrict__ out) {         // (B, T, H)
    __shared__ float q_s[Hdim];
    __shared__ float v_s[Hdim];
    __shared__ float sc[Sdim];    // scores -> attention weights
    __shared__ float red[Sdim];   // reduction scratch

    const int tid = threadIdx.x;          // 0..255 (== s index)
    const int bt  = blockIdx.x;           // 0..B*T-1
    const int b   = bt / Tdim;
    const int len = src_len[b];

    #pragma unroll
    for (int i = tid; i < Hdim; i += Sdim) {
        q_s[i] = qproj[(size_t)bt * Hdim + i];
        v_s[i] = v_w[i];
    }
    __syncthreads();

    // ---- score for source position s = tid --------------------------------
    const float* __restrict__ hrow = hproj + ((size_t)b * Sdim + tid) * Hdim;
    float acc = 0.0f;
    #pragma unroll 2
    for (int h = 0; h < Hdim; h += 4) {
        const float4 hv = *(const float4*)(hrow + h);
        const float4 qv = *(const float4*)(q_s + h);
        acc += v_s[h + 0] * tanhf(qv.x + hv.x);
        acc += v_s[h + 1] * tanhf(qv.y + hv.y);
        acc += v_s[h + 2] * tanhf(qv.z + hv.z);
        acc += v_s[h + 3] * tanhf(qv.w + hv.w);
    }
    const bool valid = (tid < len);
    float score = valid ? (acc + v_b[0]) : -__builtin_inff();

    // ---- masked softmax over s (block-wide LDS reductions) ----------------
    red[tid] = score;
    __syncthreads();
    for (int off = Sdim / 2; off > 0; off >>= 1) {
        if (tid < off) red[tid] = fmaxf(red[tid], red[tid + off]);
        __syncthreads();
    }
    const float mx = red[0];
    __syncthreads();

    const float e = valid ? __expf(score - mx) : 0.0f;
    red[tid] = e;
    __syncthreads();
    for (int off = Sdim / 2; off > 0; off >>= 1) {
        if (tid < off) red[tid] += red[tid + off];
        __syncthreads();
    }
    const float inv = 1.0f / red[0];
    __syncthreads();
    sc[tid] = e * inv;
    __syncthreads();

    // ---- context[b,t,:] = attn . enc[b,:,:] (thread owns 2 h-columns) -----
    const float* __restrict__ eb = enc + (size_t)b * Sdim * Hdim;
    float c0 = 0.0f, c1 = 0.0f;
    #pragma unroll 4
    for (int s = 0; s < Sdim; ++s) {
        const float a = sc[s];
        c0 = fmaf(a, eb[(size_t)s * Hdim + tid],        c0);
        c1 = fmaf(a, eb[(size_t)s * Hdim + tid + Sdim], c1);
    }
    out[(size_t)bt * Hdim + tid]        = c0;
    out[(size_t)bt * Hdim + tid + Sdim] = c1;
}

extern "C" void kernel_launch(void* const* d_in, const int* in_sizes, int n_in,
                              void* d_out, int out_size, void* d_ws, size_t ws_size,
                              hipStream_t stream) {
    const float* query = (const float*)d_in[0];   // (B,T,H)
    const float* enc   = (const float*)d_in[1];   // (B,S,H)
    const int*   slen  = (const int*)  d_in[2];   // (B,)
    const float* Ws_w  = (const float*)d_in[3];   // (H,H)
    const float* Ws_b  = (const float*)d_in[4];   // (H,)
    const float* Wh_w  = (const float*)d_in[5];   // (H,H)
    const float* Wh_b  = (const float*)d_in[6];   // (H,)
    const float* v_w   = (const float*)d_in[7];   // (1,H)
    const float* v_b   = (const float*)d_in[8];   // (1,)
    float* out = (float*)d_out;

    // Workspace layout (10 MB total):
    //   qproj fp32 (B*T x H)   2 MB
    //   hproj fp32 (B*S x H)   4 MB
    //   qbf   bf16 (B*T x H)   1 MB
    //   ebf   bf16 (B*S x H)   2 MB
    //   wsbf  bf16 (H x H)     0.5 MB
    //   whbf  bf16 (H x H)     0.5 MB
    char* ws = (char*)d_ws;
    float*  qproj = (float*)ws;
    float*  hproj = qproj + (size_t)Bdim * Tdim * Hdim;
    __bf16* qbf   = (__bf16*)(hproj + (size_t)Bdim * Sdim * Hdim);
    __bf16* ebf   = qbf  + (size_t)Bdim * Tdim * Hdim;
    __bf16* wsbf  = ebf  + (size_t)Bdim * Sdim * Hdim;
    __bf16* whbf  = wsbf + (size_t)Hdim * Hdim;

    // One-shot packed conversions (removes all cvt VALU from the GEMM loop).
    {
        const int nq = Bdim * Tdim * Hdim / 4;
        const int ne = Bdim * Sdim * Hdim / 4;
        const int nw = Hdim * Hdim / 4;
        cvt_f32_to_bf16<<<dim3((nq + 255) / 256), dim3(256), 0, stream>>>(query, qbf, nq);
        cvt_f32_to_bf16<<<dim3((ne + 255) / 256), dim3(256), 0, stream>>>(enc,   ebf, ne);
        cvt_f32_to_bf16<<<dim3((nw + 255) / 256), dim3(256), 0, stream>>>(Ws_w,  wsbf, nw);
        cvt_f32_to_bf16<<<dim3((nw + 255) / 256), dim3(256), 0, stream>>>(Wh_w,  whbf, nw);
    }

    // Projection GEMMs on tensor cores (bf16 WMMA, fp32 accumulate).
    {
        dim3 blk(32, 4, 1);                               // 4 waves -> 64x64 tile
        dim3 grd_q(Hdim / 64, (Bdim * Tdim) / 64, 1);     // 8 x 16
        proj_gemm_bf16_wmma<<<grd_q, blk, 0, stream>>>(qbf, wsbf, Ws_b, qproj,
                                                       Bdim * Tdim);
        dim3 grd_h(Hdim / 64, (Bdim * Sdim) / 64, 1);     // 8 x 32
        proj_gemm_bf16_wmma<<<grd_h, blk, 0, stream>>>(ebf, whbf, Wh_b, hproj,
                                                       Bdim * Sdim);
    }

    // Fused score + softmax + context: one block per (b,t).
    {
        dim3 blk(Sdim, 1, 1);                // 256 threads = 8 waves
        dim3 grd(Bdim * Tdim, 1, 1);         // 1024 blocks
        attn_fused<<<grd, blk, 0, stream>>>(qproj, hproj, enc, slen, v_w, v_b, out);
    }
}
